// BitDiscriminatorP_41652592836875
// MI455X (gfx1250) — 8-bit matrix operand order repaired and checked by reading, compiled, not run on hardware
//
#include <hip/hip_runtime.h>
#include <hip/hip_bf16.h>

typedef __attribute__((ext_vector_type(8))) int v8i;

#define NT 4

__global__ void reset_amax_kernel(unsigned* amax) {
    if (threadIdx.x == 0) *amax = 0u;
}

__global__ __launch_bounds__(256) void absmax_kernel(const float* __restrict__ x,
                                                     long long n,
                                                     unsigned* __restrict__ amax) {
    __shared__ float red[256];
    float m = 0.0f;
    long long i    = (long long)blockIdx.x * blockDim.x + threadIdx.x;
    long long strd = (long long)gridDim.x * blockDim.x;
    for (; i < n; i += strd) m = fmaxf(m, fabsf(x[i]));
    red[threadIdx.x] = m;
    __syncthreads();
    for (int s = 128; s > 0; s >>= 1) {
        if (threadIdx.x < s) red[threadIdx.x] = fmaxf(red[threadIdx.x], red[threadIdx.x + s]);
        __syncthreads();
    }
    if (threadIdx.x == 0) atomicMax(amax, __float_as_uint(red[0]));
}

__global__ __launch_bounds__(128) void quant_w_kernel(const float* __restrict__ w,
                                                      signed char* __restrict__ wq,
                                                      float* __restrict__ sw, int K) {
    const int oc = blockIdx.x;
    const float* wrow = w + (long long)oc * K;
    __shared__ float red[128];
    __shared__ float sc;
    float acc = 0.0f;
    for (int k = threadIdx.x; k < K; k += 128) acc += fabsf(wrow[k]);
    red[threadIdx.x] = acc;
    __syncthreads();
    for (int s = 64; s > 0; s >>= 1) {
        if (threadIdx.x < s) red[threadIdx.x] += red[threadIdx.x + s];
        __syncthreads();
    }
    if (threadIdx.x == 0) {
        sc = fmaxf(red[0] / (float)K, 1e-5f);
        sw[oc] = sc;
    }
    __syncthreads();
    const float inv = 1.0f / sc;
    for (int k = threadIdx.x; k < K; k += 128) {
        float q = rintf(wrow[k] * inv);
        q = fminf(fmaxf(q, -1.0f), 1.0f);
        wq[(long long)oc * K + k] = (signed char)(int)q;
    }
}

__global__ __launch_bounds__(128) void
bitconv_wmma_snake_kernel(const float* __restrict__ xin, const unsigned* __restrict__ amax,
                          const signed char* __restrict__ wq, const float* __restrict__ sw,
                          const float* __restrict__ bias, const float* __restrict__ la,
                          const float* __restrict__ lb, float* __restrict__ out,
                          int Bn, int Cin, int Hin, int Cout, int Hout,
                          int stride, int ktap, int K, int do_snake) {
    __shared__ signed char Bs[NT][16 * 64];
    __shared__ signed char As[4][16 * 64];

    const int tid    = threadIdx.x;
    const int wave   = tid >> 5;
    const int lane   = tid & 31;
    const int hi16   = lane >> 4;
    const int M      = Bn * Hout * 2;
    const int m_tile = blockIdx.x * 4 + wave;
    const int n_base = blockIdx.y * NT;

    const float s_a = fmaxf(__uint_as_float(*amax), 1e-5f);
    const float qs  = 127.0f / s_a;

    v8i acc[NT];
#pragma unroll
    for (int nt = 0; nt < NT; ++nt) acc[nt] = (v8i){0, 0, 0, 0, 0, 0, 0, 0};

    for (int k0 = 0; k0 < K; k0 += 64) {
        for (int i = tid; i < NT * 1024; i += 128) {
            const int nt = i >> 10, j = i & 1023;
            const int n = j >> 6, kk = j & 63;
            const int gn = (n_base + nt) * 16 + n;
            const int k  = k0 + kk;
            Bs[nt][j] = (gn < Cout && k < K) ? wq[(long long)gn * K + k]
                                             : (signed char)0;
        }
        for (int i = lane; i < 1024; i += 32) {
            const int m  = i >> 6, kk = i & 63;
            const int gm = m_tile * 16 + m;
            const int k  = k0 + kk;
            signed char v = 0;
            if (gm < M && k < K) {
                const int b   = gm / (Hout * 2);
                const int rem = gm - b * (Hout * 2);
                const int ho  = rem >> 1, wc = rem & 1;
                const int c   = k / ktap, t = k - c * ktap;
                const int hh  = ho * stride - 2 + t;
                if (hh >= 0 && hh < Hin) {
                    const float xv = xin[(((long long)b * Cin + c) * Hin + hh) * 2 + wc];
                    float q = rintf(xv * qs);
                    q = fminf(fmaxf(q, -128.0f), 127.0f);
                    v = (signed char)(int)q;
                }
            }
            As[wave][i] = v;
        }
        __syncthreads();

        const int* A32 = (const int*)As[wave];
        v8i afrag;
#pragma unroll
        for (int v = 0; v < 8; ++v) {
            const int am = lane & 15;
            const int ak = ((v >> 1) << 4) + (hi16 << 3) + ((v & 1) << 2);
            afrag[v] = A32[am * 16 + (ak >> 2)];
        }
#pragma unroll
        for (int nt = 0; nt < NT; ++nt) {
            const int* B32 = (const int*)Bs[nt];
            v8i bfrag;
#pragma unroll
            for (int v = 0; v < 8; ++v) {
                const int bn = lane & 15;
                const int bk = ((v >> 1) << 4) + (hi16 << 3) + ((v & 1) << 2);
                bfrag[v] = B32[bn * 16 + (bk >> 2)];
            }
            acc[nt] = __builtin_amdgcn_wmma_i32_16x16x64_iu8(true, afrag, true, bfrag,
                                                             acc[nt], false, false);
        }
        __syncthreads();
    }

#pragma unroll
    for (int nt = 0; nt < NT; ++nt) {
        const int gn = (n_base + nt) * 16 + (lane & 15);
        if (gn >= Cout) continue;
        const float dqn = (s_a / 127.0f) * sw[gn];
        const float bia = bias[gn];
        float alpha = 1.0f, invb = 0.0f;
        if (do_snake) {
            alpha = __expf(la[gn]);
            const float bb = __expf(lb[gn]);
            invb = 1.0f / (bb + 1e-9f);
        }
#pragma unroll
        for (int r = 0; r < 8; ++r) {
            const int m_loc = r + (hi16 << 3);
            const int gm    = m_tile * 16 + m_loc;
            if (gm < M) {
                const int b   = gm / (Hout * 2);
                const int rem = gm - b * (Hout * 2);
                const int ho  = rem >> 1, wc = rem & 1;
                float y = (float)acc[nt][r] * dqn + bia;
                if (do_snake) {
                    const float sn = __sinf(alpha * y);
                    y = y + invb * sn * sn;
                }
                out[(((long long)b * Cout + gn) * Hout + ho) * 2 + wc] = y;
            }
        }
    }
}

__global__ __launch_bounds__(256) void
bitconv_post_kernel(const float* __restrict__ xin, const unsigned* __restrict__ amax,
                    const signed char* __restrict__ wq, const float* __restrict__ sw,
                    const float* __restrict__ bias, float* __restrict__ fmap6,
                    float* __restrict__ outflat, int Bn, int Cin, int Hin) {
    const int Hout = Hin;
    const int gm   = blockIdx.x;
    const int b    = gm / (Hout * 2);
    const int rem  = gm - b * (Hout * 2);
    const int ho   = rem >> 1, wc = rem & 1;

    const float s_a = fmaxf(__uint_as_float(*amax), 1e-5f);
    const float qs  = 127.0f / s_a;
    const int K     = Cin * 3;

    int sum = 0;
    for (int k = threadIdx.x; k < K; k += 256) {
        const int c = k / 3, t = k - c * 3;
        const int hh = ho - 1 + t;
        if (hh >= 0 && hh < Hin) {
            const float xv = xin[(((long long)b * Cin + c) * Hin + hh) * 2 + wc];
            float q = rintf(xv * qs);
            q = fminf(fmaxf(q, -128.0f), 127.0f);
            sum += (int)q * (int)wq[k];
        }
    }
    __shared__ int red[256];
    red[threadIdx.x] = sum;
    __syncthreads();
    for (int s = 128; s > 0; s >>= 1) {
        if (threadIdx.x < s) red[threadIdx.x] += red[threadIdx.x + s];
        __syncthreads();
    }
    if (threadIdx.x == 0) {
        const float y = (float)red[0] * (s_a / 127.0f) * sw[0] + bias[0];
        fmap6[gm]   = y;
        outflat[gm] = y;
    }
}

extern "C" void kernel_launch(void* const* d_in, const int* in_sizes, int n_in,
                              void* d_out, int out_size, void* d_ws, size_t ws_size,
                              hipStream_t stream) {
    (void)in_sizes; (void)n_in; (void)out_size; (void)ws_size;

    const float* x = (const float*)d_in[0];
    const float* W[6]  = {(const float*)d_in[1], (const float*)d_in[3],
                          (const float*)d_in[5], (const float*)d_in[7],
                          (const float*)d_in[9], (const float*)d_in[11]};
    const float* Bi[6] = {(const float*)d_in[2], (const float*)d_in[4],
                          (const float*)d_in[6], (const float*)d_in[8],
                          (const float*)d_in[10], (const float*)d_in[12]};
    const float* LA[5] = {(const float*)d_in[13], (const float*)d_in[15],
                          (const float*)d_in[17], (const float*)d_in[19],
                          (const float*)d_in[21]};
    const float* LB[5] = {(const float*)d_in[14], (const float*)d_in[16],
                          (const float*)d_in[18], (const float*)d_in[20],
                          (const float*)d_in[22]};

    const int Bn = 32;
    const int Cin[6]  = {1, 32, 128, 512, 1024, 1024};
    const int Cout[6] = {32, 128, 512, 1024, 1024, 1};
    const int Hin[6]  = {32768, 10923, 3641, 1214, 405, 405};
    const int Hout[6] = {10923, 3641, 1214, 405, 405, 405};
    const int Strd[6] = {3, 3, 3, 3, 1, 1};
    const int Ktap[6] = {5, 5, 5, 5, 5, 3};

    float* outp = (float*)d_out;

    long long fmoff[6];
    long long o = (long long)Bn * Hout[5] * 2;
    for (int s = 0; s < 6; ++s) {
        fmoff[s] = o;
        o += (long long)Bn * Cout[s] * Hout[s] * 2;
    }

    unsigned*    amax = (unsigned*)d_ws;
    float*       sw   = (float*)((char*)d_ws + 64);
    signed char* wq   = (signed char*)((char*)d_ws + 8192);

    for (int s = 0; s < 6; ++s) {
        const float* in = (s == 0) ? x : (outp + fmoff[s - 1]);
        const long long n_elems = (s == 0)
            ? (long long)Bn * 1 * 65536
            : (long long)Bn * Cout[s - 1] * Hout[s - 1] * 2;
        const int K = Cin[s] * Ktap[s];

        reset_amax_kernel<<<1, 32, 0, stream>>>(amax);
        absmax_kernel<<<2048, 256, 0, stream>>>(in, n_elems, amax);
        quant_w_kernel<<<Cout[s], 128, 0, stream>>>(W[s], wq, sw, K);

        if (s < 5) {
            const int M      = Bn * Hout[s] * 2;
            const int mtiles = (M + 15) / 16;
            dim3 grid((mtiles + 3) / 4, (Cout[s] + 16 * NT - 1) / (16 * NT));
            bitconv_wmma_snake_kernel<<<grid, 128, 0, stream>>>(
                in, amax, wq, sw, Bi[s], LA[s], LB[s], outp + fmoff[s],
                Bn, Cin[s], Hin[s], Cout[s], Hout[s], Strd[s], Ktap[s], K, 1);
        } else {
            const int M = Bn * Hout[5] * 2;
            bitconv_post_kernel<<<M, 256, 0, stream>>>(
                in, amax, wq, sw, Bi[5], outp + fmoff[5], outp,
                Bn, Cin[5], Hin[5]);
        }
    }
}
